// Lstm_36936718746218
// MI455X (gfx1250) — matrix operand order repaired and checked by reading, not compiled, not run on hardware
//
#include <hip/hip_runtime.h>

// ---------------------------------------------------------------------------
// LSTM for MI455X (gfx1250): bf16 WMMA (v_wmma_f32_16x16x32_bf16), wave32.
//  Phase 0: convert emb/W_ih/W_hh to bf16, fuse biases, init h/c state.
//  Phase 1: hoisted input GEMM  gates_in[T*B,4H] = emb[x] @ W_ih^T + b  (WMMA)
//  Phase 2: 512 sequential step kernels: gates = gates_in[t] + h @ W_hh^T
//           (WMMA, W_hh bf16 = 8MB resident in 192MB L2), fused LSTM cell.
// ---------------------------------------------------------------------------

#define DEVINL __device__ __forceinline__

typedef __bf16          bfx16  __attribute__((ext_vector_type(16)));
typedef float           fx8    __attribute__((ext_vector_type(8)));
typedef unsigned short  u16;
typedef u16             u16x8  __attribute__((ext_vector_type(8)));
typedef u16             u16x16 __attribute__((ext_vector_type(16)));

constexpr int Bn = 64, Tn = 512, En = 512, Hn = 1024, Gn = 4096, Vn = 32000;

DEVINL u16 f2bf(float f) {
    unsigned u = __builtin_bit_cast(unsigned, f);
    return (u16)((u + 0x7FFFu + ((u >> 16) & 1u)) >> 16);   // round-nearest-even
}
DEVINL float bf2f(u16 v) {
    unsigned u = ((unsigned)v) << 16;
    return __builtin_bit_cast(float, u);
}

// A-fragment (16x32 bf16, M x K), rowptr = per-lane row base (row = lane%16).
// Lane layout: elems 0..15 -> k0+(lane/16)*16 .. +15 (contiguous, matches load_b).
DEVINL bfx16 load_a(const u16* rowptr, int k0) {
    int lane = threadIdx.x & 31;
    int kb = k0 + ((lane >> 4) << 4);
    u16x8 lo = *(const u16x8*)(rowptr + kb);
    u16x8 hi = *(const u16x8*)(rowptr + kb + 8);
    u16x16 c = __builtin_shufflevector(lo, hi, 0,1,2,3,4,5,6,7,8,9,10,11,12,13,14,15);
    return __builtin_bit_cast(bfx16, c);
}

// B-fragment (32x16 bf16, K x N) taken from row-major W[N][K] (B[k][n] = W[n][k]).
// Lane n = n0+lane%16 reads 16 contiguous bf16 at k0+(lane/16)*16.
DEVINL bfx16 load_b(const u16* W, int n0, int k0, int ldk) {
    int lane = threadIdx.x & 31;
    const u16* p = W + (size_t)(n0 + (lane & 15)) * ldk + k0 + ((lane >> 4) << 4);
    u16x16 c = *(const u16x16*)p;
    return __builtin_bit_cast(bfx16, c);
}

// ------------------------- Phase 0: prep / convert -------------------------
__global__ void __launch_bounds__(256) prep_kernel(
    const float* __restrict__ emb, const float* __restrict__ Wih,
    const float* __restrict__ Whh, const float* __restrict__ bih,
    const float* __restrict__ bhh, const float* __restrict__ h0,
    const float* __restrict__ c0,
    u16* __restrict__ emb_bf, u16* __restrict__ Wih_bf, u16* __restrict__ Whh_bf,
    float* __restrict__ bias, u16* __restrict__ hbf0, float* __restrict__ c)
{
    long stride = (long)gridDim.x * blockDim.x;
    long tid = (long)blockIdx.x * blockDim.x + threadIdx.x;
    for (long i = tid; i < (long)Vn * En; i += stride) emb_bf[i] = f2bf(emb[i]);
    for (long i = tid; i < (long)Gn * En; i += stride) Wih_bf[i] = f2bf(Wih[i]);
    for (long i = tid; i < (long)Gn * Hn; i += stride) Whh_bf[i] = f2bf(Whh[i]);
    for (long i = tid; i < Gn; i += stride) bias[i] = bih[i] + bhh[i];
    for (long i = tid; i < (long)Bn * Hn; i += stride) { hbf0[i] = f2bf(h0[i]); c[i] = c0[i]; }
}

// --------------------- Phase 1: hoisted input GEMM -------------------------
// gates_in[row, n] = emb[x[b,t]] . W_ih[n,:] + bias[n],  row = t*64 + b.
// Block = 256 thr (8 waves): M-tile 16 x N-tile 128. Grid (4096/128, 32768/16).
__global__ void __launch_bounds__(256) in_gemm_kernel(
    const int* __restrict__ x, const u16* __restrict__ emb_bf,
    const u16* __restrict__ Wih_bf, const float* __restrict__ bias,
    u16* __restrict__ gates_in)
{
    int wave = threadIdx.x >> 5, lane = threadIdx.x & 31;
    int n0 = blockIdx.x * 128 + wave * 16;
    int row0 = blockIdx.y * 16;
    int row = row0 + (lane & 15);
    int t = row >> 6, b = row & 63;                 // row = t*B + b
    int eidx = x[b * Tn + t];                       // x is [B,T]
    const u16* arow = emb_bf + (size_t)eidx * En;   // gathered embedding row

    fx8 acc = {};
#pragma unroll 4
    for (int k = 0; k < En; k += 32) {
        bfx16 a  = load_a(arow, k);
        bfx16 bb = load_b(Wih_bf, n0, k, En);
        acc = __builtin_amdgcn_wmma_f32_16x16x32_bf16(false, a, false, bb,
                                                      (short)0, acc, false, false);
    }
    int n = n0 + (lane & 15);
    float bv = bias[n];
#pragma unroll
    for (int r = 0; r < 8; ++r) {
        int mm = ((lane >> 4) << 3) + r;            // C layout: m = (lane/16)*8 + r
        gates_in[(size_t)(row0 + mm) * Gn + n] = f2bf(acc[r] + bv);
    }
}

// --------------------- Phase 2: recurrent step -----------------------------
// Block = 256 thr (8 waves) owns a 16(batch) x 32(h) slice and ALL 4 gates:
//   wave w -> gate (w>>1), column-half (w&1).  K-loop over H=1024 (32 WMMAs).
// Then fused LSTM cell update through LDS. Grid (32 h-tiles, 4 b-tiles).
__global__ void __launch_bounds__(256) lstm_step_kernel(
    int t, const u16* __restrict__ gates_in, const u16* __restrict__ Whh_bf,
    const u16* __restrict__ h_prev, u16* __restrict__ h_next,
    float* __restrict__ c, float* __restrict__ out)
{
    __shared__ float lds[8][16][16];
    int wave = threadIdx.x >> 5, lane = threadIdx.x & 31;
    int b0  = blockIdx.y * 16;
    int h0g = blockIdx.x * 32;
    int gate = wave >> 1, ch = wave & 1;
    int n0 = gate * Hn + h0g + ch * 16;

    const u16* arow = h_prev + (size_t)(b0 + (lane & 15)) * Hn;

    fx8 acc = {};
#pragma unroll 4
    for (int k = 0; k < Hn; k += 32) {
        bfx16 a  = load_a(arow, k);
        bfx16 bb = load_b(Whh_bf, n0, k, Hn);
        acc = __builtin_amdgcn_wmma_f32_16x16x32_bf16(false, a, false, bb,
                                                      (short)0, acc, false, false);
    }

    const u16* gin = gates_in + (size_t)t * Bn * Gn;
    int n = lane & 15;
#pragma unroll
    for (int r = 0; r < 8; ++r) {
        int mm = ((lane >> 4) << 3) + r;
        float v = acc[r] + bf2f(gin[(size_t)(b0 + mm) * Gn + n0 + n]);
        lds[wave][mm][n] = v;                       // wave = gate*2 + ch
    }
    __syncthreads();

    // 512 cell updates, 2 per thread.
#pragma unroll
    for (int e = threadIdx.x; e < 512; e += 256) {
        int bl = e >> 5, hl = e & 31;
        int chh = hl >> 4, nn = hl & 15;
        float iv = lds[0 + chh][bl][nn];
        float fv = lds[2 + chh][bl][nn];
        float gv = lds[4 + chh][bl][nn];
        float ov = lds[6 + chh][bl][nn];
        iv = 1.f / (1.f + __expf(-iv));
        fv = 1.f / (1.f + __expf(-fv));
        { float e2 = __expf(-2.f * gv); gv = (1.f - e2) / (1.f + e2); }
        ov = 1.f / (1.f + __expf(-ov));
        size_t ci = (size_t)(b0 + bl) * Hn + h0g + hl;
        float cn = fv * c[ci] + iv * gv;
        c[ci] = cn;
        float e2 = __expf(-2.f * cn);
        float hn = ov * (1.f - e2) / (1.f + e2);
        h_next[ci] = f2bf(hn);
        out[(size_t)(b0 + bl) * Tn * Hn + (size_t)t * Hn + h0g + hl] = hn;
    }
}

// ---------------------------------------------------------------------------
extern "C" void kernel_launch(void* const* d_in, const int* in_sizes, int n_in,
                              void* d_out, int out_size, void* d_ws, size_t ws_size,
                              hipStream_t stream)
{
    const int*   x    = (const int*)  d_in[0];
    const float* emb  = (const float*)d_in[1];
    const float* Wih  = (const float*)d_in[2];
    const float* Whh  = (const float*)d_in[3];
    const float* bih  = (const float*)d_in[4];
    const float* bhh  = (const float*)d_in[5];
    const float* h0   = (const float*)d_in[6];
    const float* c0   = (const float*)d_in[7];
    float* out = (float*)d_out;

    // Workspace carve-up (256B aligned). Total ~314 MB.
    char* ws = (char*)d_ws;
    size_t off = 0;
    auto carve = [&](size_t bytes) { char* p = ws + off; off = (off + bytes + 255) & ~(size_t)255; return p; };
    u16*   emb_bf   = (u16*)  carve((size_t)Vn * En * 2);       // 32.8 MB
    u16*   Wih_bf   = (u16*)  carve((size_t)Gn * En * 2);       //  4.2 MB
    u16*   Whh_bf   = (u16*)  carve((size_t)Gn * Hn * 2);       //  8.4 MB (L2-resident)
    float* bias     = (float*)carve((size_t)Gn * 4);
    u16*   hbf      = (u16*)  carve((size_t)2 * Bn * Hn * 2);   // parity double buffer
    float* cbuf     = (float*)carve((size_t)Bn * Hn * 4);
    u16*   gates_in = (u16*)  carve((size_t)Tn * Bn * Gn * 2);  // 268 MB
    (void)ws_size; (void)in_sizes; (void)n_in; (void)out_size;

    // Phase 0: convert + init state.
    prep_kernel<<<4096, 256, 0, stream>>>(emb, Wih, Whh, bih, bhh, h0, c0,
                                          emb_bf, Wih_bf, Whh_bf, bias, hbf, cbuf);

    // Phase 1: hoisted input GEMM over all (t,b).
    in_gemm_kernel<<<dim3(Gn / 128, (Bn * Tn) / 16), 256, 0, stream>>>(
        x, emb_bf, Wih_bf, bias, gates_in);

    // Phase 2: sequential recurrence, one launch per timestep.
    const size_t hstate = (size_t)Bn * Hn;
    for (int t = 0; t < Tn; ++t) {
        const u16* hp = hbf + (size_t)(t & 1) * hstate;
        u16*       hn = hbf + (size_t)((t + 1) & 1) * hstate;
        lstm_step_kernel<<<dim3(Hn / 32, Bn / 16), 256, 0, stream>>>(
            t, gates_in, Whh_bf, hp, hn, cbuf, out);
    }
}
